// ScalarTurbulenceNN_41566693490833
// MI455X (gfx1250) — matrix operand order repaired and checked by reading, not compiled, not run on hardware
//
#include <hip/hip_runtime.h>

typedef __attribute__((ext_vector_type(16))) __bf16 v16bf;
typedef __attribute__((ext_vector_type(8)))  float  v8f;

#define WMMA_BF16(a, b, c) \
  __builtin_amdgcn_wmma_f32_16x16x32_bf16(false, (a), false, (b), (short)0, (c), false, false)

__device__ __forceinline__ float sigm(float x) { return 1.0f / (1.0f + __expf(-x)); }

// A fragment: 16(M) x 32(K) bf16, LDS rows of strideB bytes, K-tile kt (32 wide).
// Layout used here (matches ldfragB): lane l -> M=l%16, h=l/16; elems 0..15 = K 16h..16h+15 contiguous
__device__ __forceinline__ v16bf ldfragA(const __bf16* buf, int strideB, int lane, int kt) {
  const char* row = (const char*)buf + (lane & 15) * strideB + kt * 64 + ((lane >> 4) << 5);
  union { v16bf v; int4 q[2]; } u;
  u.q[0] = *(const int4*)row;
  u.q[1] = *(const int4*)(row + 16);
  return u.v;
}
// B fragment: 32(K) x 16(N) bf16 from weight stored Wt[n][k]; lane l -> N=l%16, h=l/16;
// elems hold K = 16h .. 16h+15 contiguous -> one 32-byte run per lane.
__device__ __forceinline__ v16bf ldfragB(const __bf16* wt, int strideB, int lane, int nt, int kt) {
  const char* row = (const char*)wt + (nt * 16 + (lane & 15)) * strideB + kt * 64 + ((lane >> 4) << 5);
  union { v16bf v; int4 q[2]; } u;
  u.q[0] = *(const int4*)row;
  u.q[1] = *(const int4*)(row + 16);
  return u.v;
}

// ---------------- weight prep kernels (f32 -> bf16, with transpose / concat / pad) --------------
// dst[n*kstride + k] = src[k*ndim + n]  (transpose; zeros beyond kreal)
__global__ void __launch_bounds__(256) prep_transpose(const float* __restrict__ src,
                                                      __bf16* __restrict__ dst,
                                                      int ndim, int kreal, int kstride) {
  int i = blockIdx.x * 256 + threadIdx.x;
  if (i >= ndim * kstride) return;
  int n = i / kstride, k = i - n * kstride;
  dst[i] = (k < kreal) ? (__bf16)src[k * ndim + n] : (__bf16)0.0f;
}
// dst[n*kstride + k] = src[n*kdim + k]  (natural [u][v] order; zeros in pad)
__global__ void __launch_bounds__(256) prep_natural(const float* __restrict__ src,
                                                    __bf16* __restrict__ dst,
                                                    int ndim, int kdim, int kstride) {
  int i = blockIdx.x * 256 + threadIdx.x;
  if (i >= ndim * kstride) return;
  int n = i / kstride, k = i - n * kstride;
  dst[i] = (k < kdim) ? (__bf16)src[n * kdim + k] : (__bf16)0.0f;
}
// dst[n*kstride + k] = k<kA ? srcA[k*ndim+n] : (k<kTot ? srcB[(k-kA)*ndim+n] : 0)
__global__ void __launch_bounds__(256) prep_cat_t(const float* __restrict__ srcA,
                                                  const float* __restrict__ srcB,
                                                  __bf16* __restrict__ dst,
                                                  int ndim, int kA, int kTot, int kstride) {
  int i = blockIdx.x * 256 + threadIdx.x;
  if (i >= ndim * kstride) return;
  int n = i / kstride, k = i - n * kstride;
  float v = 0.0f;
  if (k < kA)        v = srcA[k * ndim + n];
  else if (k < kTot) v = srcB[(k - kA) * ndim + n];
  dst[i] = (__bf16)v;
}

// ---------------- weight region layout in d_ws / LDS (bf16 element offsets) --------------------
// W1_0t: 0      (64 x s40)     W1_1t: 2560 (32 x s40)
// per block b (base 3840 + b*20736):
//   Wa_t +0 (64 x s72)  Wb_t +4608 (32 x s40)  Wc_t +5888 (64 x s40)  Wd_t +8448 (32 x s72)
//   L0t +10752 (64 x s104, K=96 = [L0a;L0b])   L1t +17408 (32 x s104, K=96 = [L1a;L1b])
// W2_0t: 45312 (64 x s72)  R1t: 49920 (64 x s72)   total = 54528 elems = 109056 bytes
#define W_BYTES     109056
#define WAVE_LDS    29696
#define TOTAL_LDS   (W_BYTES + 4 * WAVE_LDS)   // 227840 B <= 320 KB/WGP

__global__ void __launch_bounds__(128) turbnet_kernel(
    const float* __restrict__ x0, const float* __restrict__ x1,
    const __bf16* __restrict__ wts,
    const float* __restrict__ b1_0, const float* __restrict__ Lb0g,
    const float* __restrict__ b2_0, const float* __restrict__ rb1,
    const float* __restrict__ R2g, const float* __restrict__ rb2,
    float* __restrict__ out, int nTotal)
{
  extern __shared__ char smem[];
  __bf16* wlds = (__bf16*)smem;

  // cooperative weight staging: global bf16 -> LDS (109056 B = 6816 int4)
  {
    const int4* g = (const int4*)wts;
    int4* s = (int4*)smem;
    for (int j = threadIdx.x; j < W_BYTES / 16; j += 128) s[j] = g[j];
  }
  __syncthreads();

  const int wave = threadIdx.x >> 5;
  const int lane = threadIdx.x & 31;
  const int l15  = lane & 15;
  const int hh   = lane >> 4;
  const int s0   = (blockIdx.x * 4 + wave) * 16;
  if (s0 >= nTotal) return;

  char* wb = smem + W_BYTES + wave * WAVE_LDS;
  float*  h0f   = (float*)wb;                  // [16][64]      f32
  float*  h1f   = (float*)(wb + 4096);         // [3][16][32]   f32
  __bf16* a0b   = (__bf16*)(wb + 10240);       // [16][72] bf16 (lin1/lin2 use stride 40/72)
  __bf16* a1b   = (__bf16*)(wb + 12544);       // [3][16][40] bf16
  __bf16* zbuf0 = (__bf16*)(wb + 16384);       // [16][104] bf16  ([z0a|z0b], K=96)
  __bf16* zbuf1 = (__bf16*)(wb + 19712);       // [3][16][104] bf16 ([z1a|z1b], K=96)

  // ---- load inputs as lin1 A operands (K=16 zero-padded to 32) ----
  for (int idx = lane; idx < 256; idx += 32) {
    int m = idx >> 4, u = idx & 15;
    a0b[m * 40 + u]      = (__bf16)x0[(size_t)(s0 + m) * 16 + u];
    a0b[m * 40 + 16 + u] = (__bf16)0.0f;
  }
  for (int idx = lane; idx < 768; idx += 32) {
    int m = idx / 48, r = idx - m * 48, u = r / 3, c = r - u * 3;
    a1b[(c * 16 + m) * 40 + u] = (__bf16)x1[(size_t)(s0 + m) * 48 + r];
  }
  for (int idx = lane; idx < 768; idx += 32) {
    int c = idx >> 8, m = (idx >> 4) & 15, u = idx & 15;
    a1b[(c * 16 + m) * 40 + 16 + u] = (__bf16)0.0f;
  }

  // ---- lin1: h0 = x0 @ W1_0 + b1_0 ; h1 = einsum('nui,uw->nwi', x1, W1_1) ----
  #pragma unroll
  for (int nt = 0; nt < 4; nt++) {
    v16bf A = ldfragA(a0b, 80, lane, 0);
    v16bf B = ldfragB(wlds + 0, 80, lane, nt, 0);
    v8f c = {};
    c = WMMA_BF16(A, B, c);
    float bias = b1_0[nt * 16 + l15];
    #pragma unroll
    for (int r = 0; r < 8; r++)
      h0f[(r + 8 * hh) * 64 + nt * 16 + l15] = c[r] + bias;
  }
  #pragma unroll
  for (int ci = 0; ci < 3; ci++) {
    #pragma unroll
    for (int nt = 0; nt < 2; nt++) {
      v16bf A = ldfragA(a1b + ci * 640, 80, lane, 0);
      v16bf B = ldfragB(wlds + 2560, 80, lane, nt, 0);
      v8f c = {};
      c = WMMA_BF16(A, B, c);
      #pragma unroll
      for (int r = 0; r < 8; r++)
        h1f[ci * 512 + (r + 8 * hh) * 32 + nt * 16 + l15] = c[r];
    }
  }

  // ---- two tensor-product interaction blocks ----
  for (int b = 0; b < 2; b++) {
    const __bf16* BW  = wlds + 3840 + b * 20736;
    const __bf16* Wat = BW;           // n=64 k=64 s=72
    const __bf16* Wbt = BW + 4608;    // n=32 k=32 s=40
    const __bf16* Wct = BW + 5888;    // n=64 k=32 s=40
    const __bf16* Wdt = BW + 8448;    // n=32 k=64 s=72
    const __bf16* L0t = BW + 10752;   // n=64 k=96 s=104
    const __bf16* L1t = BW + 17408;   // n=32 k=96 s=104

    // norm-activation: a0 = silu(h0); a1 = h1 * sigmoid(|h1|)
    for (int idx = lane; idx < 1024; idx += 32) {
      float v = h0f[idx];
      a0b[(idx >> 6) * 72 + (idx & 63)] = (__bf16)(v * sigm(v));
    }
    for (int idx = lane; idx < 512; idx += 32) {
      int m = idx >> 5, cc = idx & 31;
      float x = h1f[idx], y = h1f[512 + idx], z = h1f[1024 + idx];
      float g = sigm(sqrtf(x * x + y * y + z * z + 1e-8f));
      a1b[m * 40 + cc]        = (__bf16)(x * g);
      a1b[(16 + m) * 40 + cc] = (__bf16)(y * g);
      a1b[(32 + m) * 40 + cc] = (__bf16)(z * g);
    }

    // z0a[u] = a0[u] * (sum_v a0[v] Wa[u,v]) * INV64   -> zbuf0[:, 0:64]
    #pragma unroll
    for (int nt = 0; nt < 4; nt++) {
      v8f c = {};
      #pragma unroll
      for (int kt = 0; kt < 2; kt++) {
        v16bf A = ldfragA(a0b, 144, lane, kt);
        v16bf B = ldfragB(Wat, 144, lane, nt, kt);
        c = WMMA_BF16(A, B, c);
      }
      #pragma unroll
      for (int r = 0; r < 8; r++) {
        int m = r + 8 * hh, n = nt * 16 + l15;
        zbuf0[m * 104 + n] = (__bf16)(c[r] * 0.125f * (float)a0b[m * 72 + n]);
      }
    }
    // z0b[u] = sum_i a1[u,i] * (sum_v a1[v,i] Wb[u,v]) * INV96 -> zbuf0[:, 64:96]
    #pragma unroll
    for (int nt = 0; nt < 2; nt++) {
      float acc[8] = {0, 0, 0, 0, 0, 0, 0, 0};
      #pragma unroll
      for (int ci = 0; ci < 3; ci++) {
        v16bf A = ldfragA(a1b + ci * 640, 80, lane, 0);
        v16bf B = ldfragB(Wbt, 80, lane, nt, 0);
        v8f c = {};
        c = WMMA_BF16(A, B, c);
        #pragma unroll
        for (int r = 0; r < 8; r++) {
          int m = r + 8 * hh, n = nt * 16 + l15;
          acc[r] += c[r] * (float)a1b[(ci * 16 + m) * 40 + n];
        }
      }
      #pragma unroll
      for (int r = 0; r < 8; r++) {
        int m = r + 8 * hh, n = nt * 16 + l15;
        zbuf0[m * 104 + 64 + n] = (__bf16)(acc[r] * 0.10206207f);
      }
    }
    // z1a[u,i] = a0[u] * (sum_v a1[v,i] Wc[u,v]) * INV32 -> zbuf1[i][:, 0:64]
    #pragma unroll
    for (int ci = 0; ci < 3; ci++) {
      #pragma unroll
      for (int nt = 0; nt < 4; nt++) {
        v16bf A = ldfragA(a1b + ci * 640, 80, lane, 0);
        v16bf B = ldfragB(Wct, 80, lane, nt, 0);
        v8f c = {};
        c = WMMA_BF16(A, B, c);
        #pragma unroll
        for (int r = 0; r < 8; r++) {
          int m = r + 8 * hh, n = nt * 16 + l15;
          zbuf1[(ci * 16 + m) * 104 + n] = (__bf16)(c[r] * 0.17677670f * (float)a0b[m * 72 + n]);
        }
      }
    }
    // z1b[u,i] = a1[u,i] * (sum_v a0[v] Wd[u,v]) * INV64 -> zbuf1[i][:, 64:96]
    #pragma unroll
    for (int nt = 0; nt < 2; nt++) {
      v8f c = {};
      #pragma unroll
      for (int kt = 0; kt < 2; kt++) {
        v16bf A = ldfragA(a0b, 144, lane, kt);
        v16bf B = ldfragB(Wdt, 144, lane, nt, kt);
        c = WMMA_BF16(A, B, c);
      }
      #pragma unroll
      for (int r = 0; r < 8; r++) {
        int m = r + 8 * hh, n = nt * 16 + l15;
        float pre = c[r] * 0.125f;
        #pragma unroll
        for (int ci = 0; ci < 3; ci++)
          zbuf1[(ci * 16 + m) * 104 + 64 + n] = (__bf16)(pre * (float)a1b[(ci * 16 + m) * 40 + n]);
      }
    }
    // o0 = [z0a|z0b] @ [L0a;L0b] + Lb0 ; h0 += silu(o0)
    #pragma unroll
    for (int nt = 0; nt < 4; nt++) {
      v8f c = {};
      #pragma unroll
      for (int kt = 0; kt < 3; kt++) {
        v16bf A = ldfragA(zbuf0, 208, lane, kt);
        v16bf B = ldfragB(L0t, 208, lane, nt, kt);
        c = WMMA_BF16(A, B, c);
      }
      float bias = Lb0g[b * 64 + nt * 16 + l15];
      #pragma unroll
      for (int r = 0; r < 8; r++) {
        int m = r + 8 * hh, n = nt * 16 + l15;
        float o = c[r] + bias;
        h0f[m * 64 + n] += o * sigm(o);
      }
    }
    // o1[:,w,i] = [z1a|z1b](i) @ [L1a;L1b] ; gate by sigmoid(|o1|); h1 += o1
    #pragma unroll
    for (int nt = 0; nt < 2; nt++) {
      v8f c0 = {}, c1 = {}, c2 = {};
      #pragma unroll
      for (int kt = 0; kt < 3; kt++) {
        v16bf B = ldfragB(L1t, 208, lane, nt, kt);
        c0 = WMMA_BF16(ldfragA(zbuf1, 208, lane, kt), B, c0);
        c1 = WMMA_BF16(ldfragA(zbuf1 + 1664, 208, lane, kt), B, c1);
        c2 = WMMA_BF16(ldfragA(zbuf1 + 3328, 208, lane, kt), B, c2);
      }
      #pragma unroll
      for (int r = 0; r < 8; r++) {
        int m = r + 8 * hh, n = nt * 16 + l15;
        float x = c0[r], y = c1[r], z = c2[r];
        float g = sigm(sqrtf(x * x + y * y + z * z + 1e-8f));
        h1f[m * 32 + n]        += x * g;
        h1f[512 + m * 32 + n]  += y * g;
        h1f[1024 + m * 32 + n] += z * g;
      }
    }
  }

  // ---- lin2 (h0 path only; h1@W2_1 is dead code for the scalar readout) ----
  for (int idx = lane; idx < 1024; idx += 32)
    a0b[(idx >> 6) * 72 + (idx & 63)] = (__bf16)h0f[idx];
  #pragma unroll
  for (int nt = 0; nt < 4; nt++) {
    v8f c = {};
    #pragma unroll
    for (int kt = 0; kt < 2; kt++) {
      v16bf A = ldfragA(a0b, 144, lane, kt);
      v16bf B = ldfragB(wlds + 45312, 144, lane, nt, kt);
      c = WMMA_BF16(A, B, c);
    }
    float bias = b2_0[nt * 16 + l15];
    #pragma unroll
    for (int r = 0; r < 8; r++)
      h0f[(r + 8 * hh) * 64 + nt * 16 + l15] = c[r] + bias;
  }

  // ---- readout: s = silu(h0 @ R1 + rb1); out = s @ R2 + rb2 ----
  for (int idx = lane; idx < 1024; idx += 32)
    a0b[(idx >> 6) * 72 + (idx & 63)] = (__bf16)h0f[idx];
  float* sbuf = (float*)zbuf1;  // reuse: [16][64] f32
  #pragma unroll
  for (int nt = 0; nt < 4; nt++) {
    v8f c = {};
    #pragma unroll
    for (int kt = 0; kt < 2; kt++) {
      v16bf A = ldfragA(a0b, 144, lane, kt);
      v16bf B = ldfragB(wlds + 49920, 144, lane, nt, kt);
      c = WMMA_BF16(A, B, c);
    }
    float bias = rb1[nt * 16 + l15];
    #pragma unroll
    for (int r = 0; r < 8; r++) {
      float t = c[r] + bias;
      sbuf[(r + 8 * hh) * 64 + nt * 16 + l15] = t * sigm(t);
    }
  }
  // 64-wide dot with R2: 2 lanes per sample + xor-shuffle combine
  int m = lane >> 1, half = lane & 1;
  float sum = 0.0f;
  #pragma unroll 8
  for (int u = 0; u < 32; u++) {
    int uu = half * 32 + u;
    sum += sbuf[m * 64 + uu] * R2g[uu];
  }
  sum += __shfl_xor(sum, 1, 32);
  if (half == 0) out[s0 + m] = sum + rb2[0];
}

extern "C" void kernel_launch(void* const* d_in, const int* in_sizes, int n_in,
                              void* d_out, int out_size, void* d_ws, size_t ws_size,
                              hipStream_t stream) {
  const float* x0   = (const float*)d_in[0];
  const float* x1   = (const float*)d_in[1];
  const float* W1_0 = (const float*)d_in[2];
  const float* b1_0 = (const float*)d_in[3];
  const float* W1_1 = (const float*)d_in[4];
  const float* Wa   = (const float*)d_in[5];
  const float* Wb   = (const float*)d_in[6];
  const float* Wc   = (const float*)d_in[7];
  const float* Wd   = (const float*)d_in[8];
  const float* L0a  = (const float*)d_in[9];
  const float* L0b  = (const float*)d_in[10];
  const float* Lb0  = (const float*)d_in[11];
  const float* L1a  = (const float*)d_in[12];
  const float* L1b  = (const float*)d_in[13];
  const float* W2_0 = (const float*)d_in[14];
  const float* b2_0 = (const float*)d_in[15];
  const float* R1   = (const float*)d_in[17];
  const float* rb1  = (const float*)d_in[18];
  const float* R2   = (const float*)d_in[19];
  const float* rb2  = (const float*)d_in[20];
  __bf16* ws = (__bf16*)d_ws;   // needs 109056 bytes

  auto gs = [](int e) { return (e + 255) / 256; };

  // lin-layer weights: contract first index -> store transposed, K padded 16->32(40)
  prep_transpose<<<gs(64 * 40), 256, 0, stream>>>(W1_0, ws + 0,     64, 16, 40);
  prep_transpose<<<gs(32 * 40), 256, 0, stream>>>(W1_1, ws + 2560,  32, 16, 40);
  for (int b = 0; b < 2; b++) {
    __bf16* base = ws + 3840 + b * 20736;
    // 'uv' interaction weights: contract v (second index) -> natural order
    prep_natural<<<gs(64 * 72), 256, 0, stream>>>(Wa + b * 4096, base + 0,    64, 64, 72);
    prep_natural<<<gs(32 * 40), 256, 0, stream>>>(Wb + b * 1024, base + 4608, 32, 32, 40);
    prep_natural<<<gs(64 * 40), 256, 0, stream>>>(Wc + b * 2048, base + 5888, 64, 32, 40);
    prep_natural<<<gs(32 * 72), 256, 0, stream>>>(Wd + b * 2048, base + 8448, 32, 64, 72);
    // post-linear weights: contract first index -> transposed; concat K = [64|32] -> 96
    prep_cat_t<<<gs(64 * 104), 256, 0, stream>>>(L0a + b * 4096, L0b + b * 2048,
                                                 base + 10752, 64, 64, 96, 104);
    prep_cat_t<<<gs(32 * 104), 256, 0, stream>>>(L1a + b * 2048, L1b + b * 1024,
                                                 base + 17408, 32, 64, 96, 104);
  }
  prep_transpose<<<gs(64 * 72), 256, 0, stream>>>(W2_0, ws + 45312, 64, 64, 72);
  prep_transpose<<<gs(64 * 72), 256, 0, stream>>>(R1,   ws + 49920, 64, 64, 72);

  int nTotal = in_sizes[0] / 16;  // 262144
  (void)n_in; (void)out_size; (void)ws_size;

  hipFuncSetAttribute((const void*)turbnet_kernel,
                      hipFuncAttributeMaxDynamicSharedMemorySize, TOTAL_LDS);
  int grid = (nTotal + 63) / 64;  // 64 samples (4 waves x 16) per workgroup
  turbnet_kernel<<<grid, 128, TOTAL_LDS, stream>>>(
      x0, x1, ws, b1_0, Lb0, b2_0, rb1, R2, rb2, (float*)d_out, nTotal);
}
